// MixVisionMamba_73821897883828
// MI455X (gfx1250) — matrix operand order repaired and checked by reading, not compiled, not run on hardware
//
#include <hip/hip_runtime.h>
#include <math.h>
#include <stdint.h>

typedef _Float16 v16h __attribute__((ext_vector_type(16)));
typedef _Float16 v8h  __attribute__((ext_vector_type(8)));
typedef float    v8f  __attribute__((ext_vector_type(8)));

#define ISIZE  384
#define PROJW  768   // 2*ISIZE
#define ASTR   40    // padded LDS row stride (halfs): 80B, 16B-aligned, conflict-free
#define BSTR   40

__device__ __forceinline__ float siluf(float x) { return x / (1.f + expf(-x)); }
__device__ __forceinline__ float geluf(float x) { return 0.5f * x * (1.f + erff(x * 0.70710678118654752f)); }

// ----------------------------------------------------------------------------
// WMMA GEMM on pre-converted f16 operands:
//   C[M,N] = act(A[M,K] @ B[K,N] + bias) (+ C if accum)
// A: packed f16 [M,K]; BT: packed f16 [Npad,K] (pre-transposed, zero-padded).
// M % 64 == 0, K % 32 == 0, Npad % 32 == 0 guaranteed by the host -> no edge
// paths in staging. Block = 128 threads = 4 waves; tile 64(M) x 32(N);
// each wave: 16x32 via 2 WMMAs sharing one A fragment.
// Tiles staged via GLOBAL_LOAD_ASYNC_TO_LDS_B128 with double buffering:
// next tile's async copies overlap current tile's WMMAs; one
// s_wait_asynccnt 0 + one barrier per K-step.
// ----------------------------------------------------------------------------
__global__ void gemm_wmma_kernel(const _Float16* __restrict__ A, const _Float16* __restrict__ BT,
                                 const float* __restrict__ bias, float* __restrict__ C,
                                 int M, int N, int K, int ldc, int act, int accum) {
  __shared__ _Float16 As[2][64 * ASTR];   // [m][k]
  __shared__ _Float16 Bs[2][32 * BSTR];   // [n][k] (BT tile, K-contiguous)
  const int tid  = threadIdx.x;
  const int lane = tid & 31;
  const int wave = tid >> 5;
  const int blockM = blockIdx.x * 64;
  const int blockN = blockIdx.y * 32;

  v8f acc0 = {0.f, 0.f, 0.f, 0.f, 0.f, 0.f, 0.f, 0.f};
  v8f acc1 = {0.f, 0.f, 0.f, 0.f, 0.f, 0.f, 0.f, 0.f};

  // A staging: thread t -> row t>>1 (0..63), 16 halfs (32B) at (t&1)*16 -> 2x b128
  const int arow = tid >> 1;
  const int acb  = (tid & 1) << 4;
  // B staging: thread t -> row t>>2 (0..31), 8 halfs (16B) at (t&3)*8 -> 1x b128
  const int brow = tid >> 2;
  const int bcb  = (tid & 3) << 3;

  // LDS byte addresses (flat->LDS uses addr[31:0])
  const unsigned ldsA0 = (unsigned)(uintptr_t)&As[0][arow * ASTR + acb];
  const unsigned ldsA1 = (unsigned)(uintptr_t)&As[1][arow * ASTR + acb];
  const unsigned ldsB0 = (unsigned)(uintptr_t)&Bs[0][brow * BSTR + bcb];
  const unsigned ldsB1 = (unsigned)(uintptr_t)&Bs[1][brow * BSTR + bcb];

  const _Float16* aptr = A  + (size_t)(blockM + arow) * K + acb;
  const _Float16* bptr = BT + (size_t)(blockN + brow) * K + bcb;

  // Prologue: tile 0 -> buffer 0
  asm volatile(
      "global_load_async_to_lds_b128 %0, %2, off\n\t"
      "global_load_async_to_lds_b128 %1, %3, off"
      :: "v"(ldsA0), "v"(ldsA0 + 16u),
         "v"((uint64_t)(uintptr_t)aptr), "v"((uint64_t)(uintptr_t)(aptr + 8))
      : "memory");
  asm volatile(
      "global_load_async_to_lds_b128 %0, %1, off"
      :: "v"(ldsB0), "v"((uint64_t)(uintptr_t)bptr)
      : "memory");

  const int nk = K >> 5;
  for (int i = 0; i < nk; ++i) {
    asm volatile("s_wait_asynccnt 0x0" ::: "memory");
    __syncthreads();  // all waves' copies of tile i landed; prior reads of the other buffer done
    if (i + 1 < nk) {
      const _Float16* ap = aptr + (i + 1) * 32;
      const _Float16* bp = bptr + (i + 1) * 32;
      const unsigned la = (i & 1) ? ldsA0 : ldsA1;
      const unsigned lb = (i & 1) ? ldsB0 : ldsB1;
      asm volatile(
          "global_load_async_to_lds_b128 %0, %2, off\n\t"
          "global_load_async_to_lds_b128 %1, %3, off"
          :: "v"(la), "v"(la + 16u),
             "v"((uint64_t)(uintptr_t)ap), "v"((uint64_t)(uintptr_t)(ap + 8))
          : "memory");
      asm volatile(
          "global_load_async_to_lds_b128 %0, %1, off"
          :: "v"(lb), "v"((uint64_t)(uintptr_t)bp)
          : "memory");
    }

    // ---- fragments (contiguous k = 16h + s for A and B), contiguous b128 LDS reads ----
    const _Float16* Acur = As[i & 1];
    const _Float16* Bcur = Bs[i & 1];
    const int m   = lane & 15;
    const int kb  = (lane >> 4) << 4;   // A: lanes 0-15 -> K 0..15; 16-31 -> K 16..31
    const int kb2 = (lane >> 4) << 4;   // B: lanes 0-15 -> K 0..15; 16-31 -> K 16..31
    const _Float16* Ar = Acur + (wave * 16 + m) * ASTR;
    v8h a_lo = *(const v8h*)(Ar + kb);
    v8h a_hi = *(const v8h*)(Ar + kb + 8);
    v16h af = __builtin_shufflevector(a_lo, a_hi, 0,1,2,3,4,5,6,7,8,9,10,11,12,13,14,15);

    const _Float16* Br0 = Bcur + m * BSTR + kb2;
    const _Float16* Br1 = Bcur + (16 + m) * BSTR + kb2;
    v8h b0l = *(const v8h*)(Br0);
    v8h b0h = *(const v8h*)(Br0 + 8);
    v8h b1l = *(const v8h*)(Br1);
    v8h b1h = *(const v8h*)(Br1 + 8);
    v16h bf0 = __builtin_shufflevector(b0l, b0h, 0,1,2,3,4,5,6,7,8,9,10,11,12,13,14,15);
    v16h bf1 = __builtin_shufflevector(b1l, b1h, 0,1,2,3,4,5,6,7,8,9,10,11,12,13,14,15);

    acc0 = __builtin_amdgcn_wmma_f32_16x16x32_f16(false, af, false, bf0, (short)0, acc0, false, false);
    acc1 = __builtin_amdgcn_wmma_f32_16x16x32_f16(false, af, false, bf1, (short)0, acc1, false, false);
  }

  // ---- epilogue: lane n = lane&15, VGPR r -> M = r (+8 for lanes 16-31) ----
  const int n  = lane & 15;
  const int mo = blockM + wave * 16 + ((lane >> 4) << 3);
  const int gn0 = blockN + n;
  const int gn1 = blockN + 16 + n;
#pragma unroll
  for (int r = 0; r < 8; ++r) {
    int gm = mo + r;
    if (gm < M) {
      if (gn0 < N) {
        float v = acc0[r];
        if (bias) v += bias[gn0];
        if (act == 1) v = (v > 20.f) ? v : log1pf(expf(v));
        float* cp = C + (size_t)gm * ldc + gn0;
        *cp = accum ? (*cp + v) : v;
      }
      if (gn1 < N) {
        float v = acc1[r];
        if (bias) v += bias[gn1];
        if (act == 1) v = (v > 20.f) ? v : log1pf(expf(v));
        float* cp = C + (size_t)gm * ldc + gn1;
        *cp = accum ? (*cp + v) : v;
      }
    }
  }
}

// A[M,K] f32 (row stride lda) -> packed f16 [M,K]
__global__ void cvt_f16_kernel(const float* __restrict__ X, _Float16* __restrict__ Y,
                               int M, int K, int lda) {
  int idx = blockIdx.x * blockDim.x + threadIdx.x;
  if (idx >= M * K) return;
  int r = idx / K, c = idx % K;
  Y[idx] = (_Float16)X[(size_t)r * lda + c];
}

// B[K,N] f32 (row stride ldb) -> BT[Npad,K] f16, zero-padded rows N..Npad-1
__global__ void transpose_f16_kernel(const float* __restrict__ B, _Float16* __restrict__ BT,
                                     int K, int N, int ldb, int Npad) {
  int idx = blockIdx.x * blockDim.x + threadIdx.x;
  if (idx >= Npad * K) return;
  int n = idx / K, k = idx % K;
  BT[idx] = (n < N) ? (_Float16)B[(size_t)k * ldb + n] : (_Float16)0.f;
}

// ---------------------------------------------------------------------------
__global__ void layernorm_kernel(const float* __restrict__ X, float* __restrict__ Y,
                                 const float* __restrict__ g, const float* __restrict__ b,
                                 int rows, int d) {
  int row = blockIdx.x;
  if (row >= rows) return;
  const float* x = X + (size_t)row * d;
  float* y = Y + (size_t)row * d;
  __shared__ float red[256];
  float s = 0.f;
  for (int i = threadIdx.x; i < d; i += blockDim.x) s += x[i];
  red[threadIdx.x] = s; __syncthreads();
  for (int off = 128; off > 0; off >>= 1) { if (threadIdx.x < off) red[threadIdx.x] += red[threadIdx.x + off]; __syncthreads(); }
  float mu = red[0] / d; __syncthreads();
  float v = 0.f;
  for (int i = threadIdx.x; i < d; i += blockDim.x) { float t = x[i] - mu; v += t * t; }
  red[threadIdx.x] = v; __syncthreads();
  for (int off = 128; off > 0; off >>= 1) { if (threadIdx.x < off) red[threadIdx.x] += red[threadIdx.x + off]; __syncthreads(); }
  float rstd = rsqrtf(red[0] / d + 1e-5f);
  for (int i = threadIdx.x; i < d; i += blockDim.x) y[i] = (x[i] - mu) * rstd * g[i] + b[i];
}

// Patch-embedding conv (strided, padded); input NCHW (stage 0) or NHWC
__global__ void patch_conv_kernel(const float* __restrict__ X, const float* __restrict__ Wt,
                                  const float* __restrict__ bias, float* __restrict__ Y,
                                  int Hin, int Win, int Cin, int Hout, int Wout, int Cout,
                                  int k, int strd, int pad, int in_nchw) {
  int idx = blockIdx.x * blockDim.x + threadIdx.x;
  int total = Hout * Wout * Cout;
  if (idx >= total) return;
  int co = idx % Cout;
  int wo = (idx / Cout) % Wout;
  int ho = idx / (Cout * Wout);
  float s = bias[co];
  for (int ky = 0; ky < k; ++ky) {
    int hi = ho * strd - pad + ky;
    if (hi < 0 || hi >= Hin) continue;
    for (int kx = 0; kx < k; ++kx) {
      int wi = wo * strd - pad + kx;
      if (wi < 0 || wi >= Win) continue;
      for (int ci = 0; ci < Cin; ++ci) {
        float xv = in_nchw ? X[(size_t)ci * Hin * Win + (size_t)hi * Win + wi]
                           : X[((size_t)hi * Win + wi) * Cin + ci];
        s += xv * Wt[(((size_t)ky * k + kx) * Cin + ci) * Cout + co];
      }
    }
  }
  Y[idx] = s;
}

__global__ void inv_perm_kernel(const int* __restrict__ perm, int* __restrict__ inv, int L) {
  int l = blockIdx.x * blockDim.x + threadIdx.x;
  if (l < L) inv[perm[l]] = l;
}

// XI = concat(x, x reversed, x gathered by perm) over stream dim
__global__ void build_streams_kernel(const float* __restrict__ X, float* __restrict__ XI,
                                     const int* __restrict__ perm, int L, int d) {
  int idx = blockIdx.x * blockDim.x + threadIdx.x;
  if (idx >= L * d) return;
  int c = idx % d, l = idx / d;
  float v = X[idx];
  XI[idx] = v;
  XI[((size_t)(L + (L - 1 - l))) * d + c] = v;
  XI[((size_t)(2 * L + l)) * d + c] = X[(size_t)perm[l] * d + c];
}

// causal conv1d over h = PROJ[:, :, 0:384], then SiLU
__global__ void causal_conv_silu_kernel(const float* __restrict__ PROJ, const float* __restrict__ cw,
                                        const float* __restrict__ cb, float* __restrict__ OUT,
                                        int SL, int L) {
  int idx = blockIdx.x * blockDim.x + threadIdx.x;
  if (idx >= SL * ISIZE) return;
  int i = idx % ISIZE;
  int t = idx / ISIZE;
  int b = t / L, l = t % L;
  float s = cb[i];
#pragma unroll
  for (int k = 0; k < 4; ++k) {
    int lk = l - 3 + k;
    if (lk < 0) continue;
    s += PROJ[((size_t)(b * L + lk)) * PROJW + i] * cw[i * 4 + k];
  }
  OUT[idx] = siluf(s);
}

// Selective scan: one thread per (stream, channel); dt buffer overwritten with y
__global__ void scan_kernel(float* __restrict__ DT, const float* __restrict__ SSM,
                            const float* __restrict__ CONVH, const float* __restrict__ PROJ,
                            const float* __restrict__ A_log, const float* __restrict__ Dp,
                            int L, int d, int ssmld) {
  int idx = blockIdx.x * blockDim.x + threadIdx.x;
  if (idx >= 3 * ISIZE) return;
  int b = idx / ISIZE, i = idx % ISIZE;
  float Ai[4];
#pragma unroll
  for (int n = 0; n < 4; ++n) Ai[n] = -expf(A_log[i * 4 + n]);
  float Di = Dp[i];
  float h[4] = {0.f, 0.f, 0.f, 0.f};
  for (int l = 0; l < L; ++l) {
    size_t row = (size_t)(b * L + l);
    float dt = DT[row * ISIZE + i];
    const float* Bm = SSM + row * ssmld + d;
    const float* Cm = Bm + 4;
    float u = CONVH[row * ISIZE + i];
    float y = 0.f;
#pragma unroll
    for (int n = 0; n < 4; ++n) {
      float dA = expf(dt * Ai[n]);
      h[n] = dA * h[n] + dt * Bm[n] * u;
      y += h[n] * Cm[n];
    }
    y += u * Di;
    float g = PROJ[row * PROJW + ISIZE + i];
    y *= siluf(g);
    DT[row * ISIZE + i] = y;
  }
}

// mean over L per (stream, channel) — reversal/permutation invariant
__global__ void gate_means_kernel(const float* __restrict__ MOUT, float* __restrict__ GIN,
                                  int L, int d) {
  int idx = blockIdx.x * blockDim.x + threadIdx.x;
  if (idx >= 3 * d) return;
  int s = idx / d, c = idx % d;
  float acc = 0.f;
  for (int l = 0; l < L; ++l) acc += MOUT[((size_t)(s * L + l)) * d + c];
  GIN[idx] = acc / (float)L;
}

__global__ void gate_softmax_kernel(const float* __restrict__ GIN, const float* __restrict__ gw,
                                    float* __restrict__ G, int d3) {
  __shared__ float lg[3];
  if (threadIdx.x < 3) {
    float s = 0.f;
    for (int c = 0; c < d3; ++c) s += GIN[c] * gw[c * 3 + threadIdx.x];
    lg[threadIdx.x] = s;
  }
  __syncthreads();
  if (threadIdx.x == 0) {
    float m = fmaxf(lg[0], fmaxf(lg[1], lg[2]));
    float e0 = expf(lg[0] - m), e1 = expf(lg[1] - m), e2 = expf(lg[2] - m);
    float sum = e0 + e1 + e2;
    G[0] = e0 / sum; G[1] = e1 / sum; G[2] = e2 / sum;
  }
}

__global__ void combine_kernel(float* __restrict__ X, const float* __restrict__ MOUT,
                               const float* __restrict__ G, const int* __restrict__ inv,
                               int L, int d) {
  int idx = blockIdx.x * blockDim.x + threadIdx.x;
  if (idx >= L * d) return;
  int c = idx % d, l = idx / d;
  float f = MOUT[(size_t)l * d + c];
  float r = MOUT[((size_t)(L + (L - 1 - l))) * d + c];
  float s = MOUT[((size_t)(2 * L + inv[l])) * d + c];
  X[idx] += G[0] * f + G[1] * r + G[2] * s;
}

// depthwise 3x3 SAME conv + exact GELU
__global__ void dwconv_gelu_kernel(const float* __restrict__ Hin, const float* __restrict__ pw,
                                   const float* __restrict__ pb, float* __restrict__ Out,
                                   int H, int W, int f) {
  int idx = blockIdx.x * blockDim.x + threadIdx.x;
  if (idx >= H * W * f) return;
  int c = idx % f;
  int w = (idx / f) % W;
  int h = idx / (f * W);
  float s = pb[c];
#pragma unroll
  for (int ky = 0; ky < 3; ++ky) {
    int hy = h - 1 + ky;
    if (hy < 0 || hy >= H) continue;
#pragma unroll
    for (int kx = 0; kx < 3; ++kx) {
      int wx = w - 1 + kx;
      if (wx < 0 || wx >= W) continue;
      s += Hin[((size_t)hy * W + wx) * f + c] * pw[(ky * 3 + kx) * f + c];
    }
  }
  Out[idx] = geluf(s);
}

__global__ void write_nchw_kernel(const float* __restrict__ Xn, float* __restrict__ Out,
                                  int H, int W, int C) {
  int idx = blockIdx.x * blockDim.x + threadIdx.x;
  if (idx >= H * W * C) return;
  int c = idx % C;
  int w = (idx / C) % W;
  int h = idx / (C * W);
  Out[(size_t)c * H * W + (size_t)h * W + w] = Xn[idx];
}

// ---------------------------------------------------------------------------
extern "C" void kernel_launch(void* const* d_in, const int* in_sizes, int n_in,
                              void* d_out, int out_size, void* d_ws, size_t ws_size,
                              hipStream_t stream) {
  (void)in_sizes; (void)n_in; (void)out_size; (void)ws_size;
  const int NUM_LAYERS[4] = {3, 4, 6, 3};
  const int HEADS[4]      = {1, 2, 4, 8};
  const int PATCH[4]      = {7, 3, 3, 3};
  const int STRD[4]       = {4, 2, 2, 2};

  int ip = 0;
  const float* x_in = (const float*)d_in[ip++];

  const float* SP[4][4];
  const float* LP[16][20];
  int li = 0;
  for (int s = 0; s < 4; ++s) {
    for (int j = 0; j < 4; ++j) SP[s][j] = (const float*)d_in[ip++];
    for (int l = 0; l < NUM_LAYERS[s]; ++l, ++li)
      for (int j = 0; j < 20; ++j) LP[li][j] = (const float*)d_in[ip++];
  }
  const int* PERM[16];
  for (int p = 0; p < 16; ++p) PERM[p] = (const int*)d_in[ip++];

  float* W = (float*)d_ws;
  size_t off = 0;
  auto alloc = [&](size_t n) { float* p = W + off; off += n; return p; };
  float* XBUF  = alloc(262144);   // current activation [L, d]
  float* XN    = alloc(262144);   // stage-normalized activation
  float* XI    = alloc(786432);   // 3 streams [3L, d]
  float* H1B   = alloc(786432);   // ln1(xi)
  float* PROJ  = alloc(9437184);  // [3L, 768]
  float* CONVH = alloc(4718592);  // [3L, 384]
  float* SSMB  = alloc(884736);   // [3L, d+8]
  float* DTB   = alloc(4718592);  // dt -> (in-place) y  [3L, 384]
  float* MOUT  = alloc(786432);   // mamba out [3L, d]
  float* LN2B  = alloc(262144);
  float* FF1   = alloc(1048576);  // [L, 4d]
  float* FF2   = alloc(1048576);
  float* GIN   = alloc(2048);
  float* G     = alloc(8);
  int*   INV   = (int*)alloc(4096);
  _Float16* A16  = (_Float16*)alloc(2359296);  // 4,718,592 halfs: packed f16 A
  _Float16* BT16 = (_Float16*)alloc(524288);   // 1,048,576 halfs: f16 B^T [Npad,K]

  auto cdiv = [](int a, int b) { return (a + b - 1) / b; };
  auto gemm = [&](const float* A, const float* B, const float* bias, float* C,
                  int M, int N, int K, int lda, int ldb, int ldc, int act, int accum) {
    int Npad = cdiv(N, 32) * 32;
    cvt_f16_kernel<<<cdiv(M * K, 256), 256, 0, stream>>>(A, A16, M, K, lda);
    transpose_f16_kernel<<<cdiv(Npad * K, 256), 256, 0, stream>>>(B, BT16, K, N, ldb, Npad);
    dim3 grid(M / 64, Npad / 32);
    gemm_wmma_kernel<<<grid, dim3(128), 0, stream>>>(A16, BT16, bias, C, M, N, K, ldc, act, accum);
  };

  int Hc = 256, Wc = 256, Cin = 3;
  const float* Xsrc = x_in;
  int in_nchw = 1;
  size_t out_off = 0;
  int layer_idx = 0;

  for (int s = 0; s < 4; ++s) {
    int d = 64 * HEADS[s];
    int f = 4 * d;
    int k = PATCH[s], strd = STRD[s], pad = k / 2;
    int Ho = (Hc + 2 * pad - k) / strd + 1;
    int Wo = (Wc + 2 * pad - k) / strd + 1;
    int L = Ho * Wo;

    patch_conv_kernel<<<cdiv(L * d, 256), 256, 0, stream>>>(
        Xsrc, SP[s][0], SP[s][1], XBUF, Hc, Wc, Cin, Ho, Wo, d, k, strd, pad, in_nchw);

    for (int l = 0; l < NUM_LAYERS[s]; ++l, ++layer_idx) {
      const float* const* p = LP[layer_idx];
      // 0 ln1_g 1 ln1_b 2 ln2_g 3 ln2_b 4 in_w 5 conv_w 6 conv_b 7 x_w 8 dt_w 9 dt_b
      // 10 A_log 11 D 12 out_w 13 fc1_w 14 fc1_b 15 pe_w 16 pe_b 17 fc2_w 18 fc2_b 19 gate_w
      inv_perm_kernel<<<cdiv(L, 256), 256, 0, stream>>>(PERM[layer_idx], INV, L);
      build_streams_kernel<<<cdiv(L * d, 256), 256, 0, stream>>>(XBUF, XI, PERM[layer_idx], L, d);
      layernorm_kernel<<<3 * L, 256, 0, stream>>>(XI, H1B, p[0], p[1], 3 * L, d);
      gemm(H1B, p[4], nullptr, PROJ, 3 * L, PROJW, d, d, PROJW, PROJW, 0, 0);
      causal_conv_silu_kernel<<<cdiv(3 * L * ISIZE, 256), 256, 0, stream>>>(PROJ, p[5], p[6], CONVH, 3 * L, L);
      int nssm = d + 8;
      gemm(CONVH, p[7], nullptr, SSMB, 3 * L, nssm, ISIZE, ISIZE, nssm, nssm, 0, 0);
      gemm(SSMB, p[8], p[9], DTB, 3 * L, ISIZE, d, nssm, ISIZE, ISIZE, /*softplus*/1, 0);
      scan_kernel<<<cdiv(3 * ISIZE, 256), 256, 0, stream>>>(DTB, SSMB, CONVH, PROJ, p[10], p[11], L, d, nssm);
      gemm(DTB, p[12], nullptr, MOUT, 3 * L, d, ISIZE, ISIZE, d, d, 0, 0);
      gate_means_kernel<<<cdiv(3 * d, 256), 256, 0, stream>>>(MOUT, GIN, L, d);
      gate_softmax_kernel<<<1, 32, 0, stream>>>(GIN, p[19], G, 3 * d);
      combine_kernel<<<cdiv(L * d, 256), 256, 0, stream>>>(XBUF, MOUT, G, INV, L, d);
      layernorm_kernel<<<L, 256, 0, stream>>>(XBUF, LN2B, p[2], p[3], L, d);
      gemm(LN2B, p[13], p[14], FF1, L, f, d, d, f, f, 0, 0);
      dwconv_gelu_kernel<<<cdiv(L * f, 256), 256, 0, stream>>>(FF1, p[15], p[16], FF2, Ho, Wo, f);
      gemm(FF2, p[17], p[18], XBUF, L, d, f, f, d, d, 0, /*accum residual*/1);
    }

    layernorm_kernel<<<L, 256, 0, stream>>>(XBUF, XN, SP[s][2], SP[s][3], L, d);
    write_nchw_kernel<<<cdiv(L * d, 256), 256, 0, stream>>>(XN, (float*)d_out + out_off, Ho, Wo, d);
    out_off += (size_t)L * d;

    Xsrc = XN; in_nchw = 0; Hc = Ho; Wc = Wo; Cin = d;
  }
}
